// MultiHeadAttention_22282290331664
// MI455X (gfx1250) — matrix operand order repaired and checked by reading, not compiled, not run on hardware
//
#include <hip/hip_runtime.h>
#include <hip/hip_bf16.h>

// MI455X / gfx1250 fused MHA forward.
// Path: 3x projection GEMM (fp32 -> bf16 WMMA, fp32 accum, bf16 out to ws;
//       log2(e)/sqrt(DH) folded into the Q projection so softmax runs in
//       the base-2 domain on raw v_exp_f32)
//       -> flash attention (bf16 WMMA QK^T + PV, fp32 online softmax with
//          DPP16 butterfly reductions, double-buffered K/V tiles)
//       -> output projection (bf16 WMMA, fp32 out).
// Workspace: 4 * 4096*1024 bf16 buffers = 33,554,432 bytes.

typedef __bf16 bf16;
typedef __attribute__((ext_vector_type(16))) __bf16 v16bf;
typedef __attribute__((ext_vector_type(8)))  __bf16 v8bf;
typedef __attribute__((ext_vector_type(8)))  float  v8f;

#define WMMA_BF16(a, b, c)                                                    \
  __builtin_amdgcn_wmma_f32_16x16x32_bf16(false, (a), false, (b), (short)0,   \
                                          (c), false, false)

// Raw hardware base-2 exponential (v_exp_f32), no log2e pre-multiply.
#if __has_builtin(__builtin_amdgcn_exp2f)
#define EXP2F(x) __builtin_amdgcn_exp2f(x)
#else
#define EXP2F(x) exp2f(x)
#endif

// ---------------------------------------------------------------------------
// 16-lane butterfly reductions in pure VALU via DPP16 (no DS pipe traffic).
// xor1 = quad_perm[1,0,3,2]=0xB1, xor2 = quad_perm[2,3,0,1]=0x4E,
// then row_half_mirror (0x141) crosses quads, row_mirror (0x140) crosses the
// two 8-groups of a 16-lane row. Lanes 0-15 / 16-31 reduce independently.
// ---------------------------------------------------------------------------
#if __has_builtin(__builtin_amdgcn_mov_dpp)
#define DPP_STEP(ui, ctrl)                                                    \
  __builtin_amdgcn_mov_dpp((ui), (ctrl), 0xF, 0xF, true)
static __device__ __forceinline__ float red16_max(float x) {
  union { float f; int i; } u, t;
  u.f = x;
  t.i = DPP_STEP(u.i, 0xB1);  u.f = fmaxf(u.f, t.f);
  t.i = DPP_STEP(u.i, 0x4E);  u.f = fmaxf(u.f, t.f);
  t.i = DPP_STEP(u.i, 0x141); u.f = fmaxf(u.f, t.f);
  t.i = DPP_STEP(u.i, 0x140); u.f = fmaxf(u.f, t.f);
  return u.f;
}
static __device__ __forceinline__ float red16_sum(float x) {
  union { float f; int i; } u, t;
  u.f = x;
  t.i = DPP_STEP(u.i, 0xB1);  u.f += t.f;
  t.i = DPP_STEP(u.i, 0x4E);  u.f += t.f;
  t.i = DPP_STEP(u.i, 0x141); u.f += t.f;
  t.i = DPP_STEP(u.i, 0x140); u.f += t.f;
  return u.f;
}
#else
static __device__ __forceinline__ float red16_max(float x) {
  x = fmaxf(x, __shfl_xor(x, 1)); x = fmaxf(x, __shfl_xor(x, 2));
  x = fmaxf(x, __shfl_xor(x, 4)); x = fmaxf(x, __shfl_xor(x, 8));
  return x;
}
static __device__ __forceinline__ float red16_sum(float x) {
  x += __shfl_xor(x, 1); x += __shfl_xor(x, 2);
  x += __shfl_xor(x, 4); x += __shfl_xor(x, 8);
  return x;
}
#endif

// ---------------------------------------------------------------------------
// Fragment loaders (LDS -> VGPR), layouts per CDNA5 ISA §7.12.2
// ---------------------------------------------------------------------------

// A-matrix 16x32 bf16, row-major LDS tile, leading dim ldk (elements).
// lane L: row = row_base + L%16; elems 0..7 = K[k0..k0+7], 8..15 = K[k0+16..],
// k0 = kbase + (L/16)*8.
static __device__ __forceinline__ v16bf load_a_frag(const bf16* s, int row_base,
                                                    int ldk, int kbase, int lane) {
  const bf16* p = s + (row_base + (lane & 15)) * ldk + kbase + ((lane >> 4) << 3);
  v8bf lo = *(const v8bf*)(p);
  v8bf hi = *(const v8bf*)(p + 16);
  v16bf r;
#pragma unroll
  for (int i = 0; i < 8; ++i) { r[i] = lo[i]; r[i + 8] = hi[i]; }
  return r;
}

// B-matrix 32x16 bf16. Column n of B is stored as row (col_base + n) of the
// LDS tile (K-contiguous). lane L: col = col_base + L%16; elems 0..7 =
// K[k0..k0+7], 8..15 = K[k0+16..], k0 = kbase + (L/16)*8.
static __device__ __forceinline__ v16bf load_b_frag(const bf16* s, int col_base,
                                                    int ldk, int kbase, int lane) {
  const bf16* p = s + (col_base + (lane & 15)) * ldk + kbase + ((lane >> 4) << 3);
  v8bf lo = *(const v8bf*)(p);
  v8bf hi = *(const v8bf*)(p + 16);
  v16bf r;
#pragma unroll
  for (int i = 0; i < 8; ++i) { r[i] = lo[i]; r[i + 8] = hi[i]; }
  return r;
}

// ---------------------------------------------------------------------------
// NT GEMM: Y[r, n] = oscale * sum_k X[r, k] * W[n, k]   (i.e. X @ W.T)
// X: [4096,1024] (fp32 or bf16), W: [1024,1024] fp32, Y: [4096,1024].
// Block tile 128(M) x 128(N), K-step 32, double-buffered LDS, register-staged
// global loads so VMEM overlaps WMMA. 8 waves: (w%4)*32 rows x (w/4)*64 cols.
// ---------------------------------------------------------------------------
template <bool IN_BF16, bool OUT_BF16>
__global__ __launch_bounds__(256) void gemm_nt_kernel(
    const void* __restrict__ Xv, const float* __restrict__ W,
    void* __restrict__ Yv, float oscale) {
  constexpr int LD = 1024;
  constexpr int KT = 32;
  constexpr int NK = LD / KT;
  __shared__ __align__(16) bf16 Xs[2][128 * KT];
  __shared__ __align__(16) bf16 Ws[2][128 * KT];

  const int tid = threadIdx.x;
  const int lane = tid & 31;
  const int wv = tid >> 5;
  const int rbase = blockIdx.y * 128;
  const int cbase = blockIdx.x * 128;
  const int wm = (wv & 3) * 32;
  const int wn = (wv >> 2) * 64;

  const v8f vzero = {0.f, 0.f, 0.f, 0.f, 0.f, 0.f, 0.f, 0.f};
  v8f acc[2][4];
#pragma unroll
  for (int i = 0; i < 2; ++i)
#pragma unroll
    for (int j = 0; j < 4; ++j) acc[i][j] = vzero;

  float4 xr[4], wr[4];   // fp32 staging (1024 f4 / 256 thr = 4 each)
  v8bf   xb[2];          // bf16 staging (512 v8bf / 256 thr = 2 each)

  // ---- global load of tile kb into registers ----
  auto g_load = [&](int kb) {
    if (IN_BF16) {
      const bf16* X = (const bf16*)Xv;
#pragma unroll
      for (int i = 0; i < 2; ++i) {
        int e = tid + i * 256, r = e >> 2, c = (e & 3) << 3;
        xb[i] = *(const v8bf*)(X + (size_t)(rbase + r) * LD + kb + c);
      }
    } else {
      const float* X = (const float*)Xv;
#pragma unroll
      for (int i = 0; i < 4; ++i) {
        int e = tid + i * 256, r = e >> 3, c = (e & 7) << 2;
        xr[i] = *(const float4*)(X + (size_t)(rbase + r) * LD + kb + c);
      }
    }
#pragma unroll
    for (int i = 0; i < 4; ++i) {
      int e = tid + i * 256, r = e >> 3, c = (e & 7) << 2;
      wr[i] = *(const float4*)(W + (size_t)(cbase + r) * LD + kb + c);
    }
  };
  // ---- register -> LDS (bf16 convert for fp32 inputs) ----
  auto s_store = [&](int buf) {
    if (IN_BF16) {
#pragma unroll
      for (int i = 0; i < 2; ++i) {
        int e = tid + i * 256, r = e >> 2, c = (e & 3) << 3;
        *(v8bf*)&Xs[buf][r * KT + c] = xb[i];
      }
    } else {
#pragma unroll
      for (int i = 0; i < 4; ++i) {
        int e = tid + i * 256, r = e >> 3, c = (e & 7) << 2;
        bf16* d = &Xs[buf][r * KT + c];
        d[0] = (bf16)xr[i].x; d[1] = (bf16)xr[i].y;
        d[2] = (bf16)xr[i].z; d[3] = (bf16)xr[i].w;
      }
    }
#pragma unroll
    for (int i = 0; i < 4; ++i) {
      int e = tid + i * 256, r = e >> 3, c = (e & 7) << 2;
      bf16* d = &Ws[buf][r * KT + c];
      d[0] = (bf16)wr[i].x; d[1] = (bf16)wr[i].y;
      d[2] = (bf16)wr[i].z; d[3] = (bf16)wr[i].w;
    }
  };

  g_load(0);
  s_store(0);
  for (int kbi = 0; kbi < NK; ++kbi) {
    const int cur = kbi & 1;
    __syncthreads();  // buf[cur] complete; prior reads of buf[cur^1] done
    if (kbi + 1 < NK) g_load((kbi + 1) * KT);

    v16bf a0 = load_a_frag(Xs[cur], wm, KT, 0, lane);
    v16bf a1 = load_a_frag(Xs[cur], wm + 16, KT, 0, lane);
#pragma unroll
    for (int j = 0; j < 4; ++j) {
      v16bf b = load_b_frag(Ws[cur], wn + j * 16, KT, 0, lane);
      acc[0][j] = WMMA_BF16(a0, b, acc[0][j]);
      acc[1][j] = WMMA_BF16(a1, b, acc[1][j]);
    }
    if (kbi + 1 < NK) s_store(cur ^ 1);
  }

  // --- epilogue: C/D layout row = e + (lane/16)*8, col = lane%16 ---
  const int hh = lane >> 4, ln = lane & 15;
#pragma unroll
  for (int i = 0; i < 2; ++i)
#pragma unroll
    for (int j = 0; j < 4; ++j)
#pragma unroll
      for (int e = 0; e < 8; ++e) {
        size_t row = rbase + wm + i * 16 + hh * 8 + e;
        size_t col = cbase + wn + j * 16 + ln;
        float vv = acc[i][j][e] * oscale;
        if (OUT_BF16)
          ((bf16*)Yv)[row * LD + col] = (bf16)vv;
        else
          ((float*)Yv)[row * LD + col] = vv;
      }
}

// ---------------------------------------------------------------------------
// Flash attention over projected bf16 Q/K/V laid out [B*S, D], head h at
// column h*64. Q carries log2(e)/sqrt(DH), so scores are already in the
// base-2 domain and softmax uses raw v_exp_f32. One block = 128 q-rows of
// one (b, h); 8 waves, 16 q-rows each. Double-buffered K/V tiles.
// ---------------------------------------------------------------------------
__global__ __launch_bounds__(256) void flash_attn_kernel(
    const bf16* __restrict__ Qp, const bf16* __restrict__ Kp,
    const bf16* __restrict__ Vp, bf16* __restrict__ Out) {
  constexpr int S = 2048, Dm = 1024, DH = 64, BQ = 128, BK = 64;
  constexpr int NKB = S / BK;
  __shared__ __align__(16) bf16 Qs[BQ * DH];         // 16 KB
  __shared__ __align__(16) bf16 Ks[2][BK * DH];      // 16 KB (rows = K rows)
  __shared__ __align__(16) bf16 Vt[2][DH * BK];      // 16 KB Vt[d][kk]=V[kk][d]
  __shared__ __align__(16) bf16 Ps[8][16 * DH];      // 16 KB per-wave P buffer

  const int tid = threadIdx.x;
  const int lane = tid & 31;
  const int wv = tid >> 5;
  const int hh = lane >> 4, ln = lane & 15;
  const int b = blockIdx.y >> 4;
  const int h = blockIdx.y & 15;
  const int q0 = blockIdx.x * BQ;

  const bf16* Qg = Qp + (size_t)(b * S + q0) * Dm + h * DH;
  const bf16* Kg = Kp + (size_t)(b * S) * Dm + h * DH;
  const bf16* Vg = Vp + (size_t)(b * S) * Dm + h * DH;

  // stage Q tile (128 x 64) once
#pragma unroll
  for (int i = 0; i < 4; ++i) {
    int e = tid + i * 256, r = e >> 3, c = (e & 7) << 3;
    *(v8bf*)&Qs[r * DH + c] = *(const v8bf*)(Qg + (size_t)r * Dm + c);
  }

  v8bf kreg[2], vreg[2];  // register staging: 512 v8bf / 256 thr = 2 each
  auto g_load = [&](int kb) {
#pragma unroll
    for (int i = 0; i < 2; ++i) {
      int e = tid + i * 256, r = e >> 3, c = (e & 7) << 3;
      kreg[i] = *(const v8bf*)(Kg + (size_t)(kb + r) * Dm + c);
      vreg[i] = *(const v8bf*)(Vg + (size_t)(kb + r) * Dm + c);
    }
  };
  auto s_store = [&](int buf) {
#pragma unroll
    for (int i = 0; i < 2; ++i) {
      int e = tid + i * 256, r = e >> 3, c = (e & 7) << 3;
      *(v8bf*)&Ks[buf][r * DH + c] = kreg[i];
#pragma unroll
      for (int t = 0; t < 8; ++t) Vt[buf][(c + t) * BK + r] = vreg[i][t];
    }
  };

  const v8f vzero = {0.f, 0.f, 0.f, 0.f, 0.f, 0.f, 0.f, 0.f};
  v8f o[4];
  float m[8], l[8];
#pragma unroll
  for (int j = 0; j < 4; ++j) o[j] = vzero;
#pragma unroll
  for (int e = 0; e < 8; ++e) { m[e] = -1e30f; l[e] = 0.f; }

  const int qr = wv * 16;

  g_load(0);
  s_store(0);
  for (int kbi = 0; kbi < NKB; ++kbi) {
    const int cur = kbi & 1;
    __syncthreads();  // buf[cur] ready; prior reads of buf[cur^1] finished
    if (kbi + 1 < NKB) g_load((kbi + 1) * BK);

    // scores: 16 x 64 = Q(16x64) x K^T, K-dim = DH in two WMMA steps
    v16bf a0 = load_a_frag(Qs, qr, DH, 0, lane);
    v16bf a1 = load_a_frag(Qs, qr, DH, 32, lane);
    v8f s[4];
#pragma unroll
    for (int j = 0; j < 4; ++j) {
      v16bf b0 = load_b_frag(Ks[cur], j * 16, DH, 0, lane);
      s[j] = WMMA_BF16(a0, b0, vzero);
      v16bf b1 = load_b_frag(Ks[cur], j * 16, DH, 32, lane);
      s[j] = WMMA_BF16(a1, b1, s[j]);
    }

    // online softmax in base-2 domain; row r = e + hh*8 spans 16 lanes
#pragma unroll
    for (int e = 0; e < 8; ++e) {
      float mv = fmaxf(fmaxf(s[0][e], s[1][e]), fmaxf(s[2][e], s[3][e]));
      mv = red16_max(mv);
      float mn = fmaxf(m[e], mv);
      float alpha = EXP2F(m[e] - mn);
      m[e] = mn;
      float ps = 0.f;
#pragma unroll
      for (int j = 0; j < 4; ++j) {
        float p = EXP2F(s[j][e] - mn);
        s[j][e] = p;
        ps += p;
      }
      l[e] = l[e] * alpha + red16_sum(ps);
#pragma unroll
      for (int j = 0; j < 4; ++j) o[j][e] *= alpha;
    }

    // re-shape P from C-layout to A-layout via per-wave LDS buffer
    bf16* Pw = Ps[wv];
#pragma unroll
    for (int j = 0; j < 4; ++j)
#pragma unroll
      for (int e = 0; e < 8; ++e)
        Pw[(hh * 8 + e) * DH + j * 16 + ln] = (bf16)s[j][e];
    __builtin_amdgcn_wave_barrier();  // same-wave LDS ops are in-order

    // O(16x64) += P(16x64) x V(64x64); B-frag columns d come from Vt rows
#pragma unroll
    for (int ks = 0; ks < 2; ++ks) {
      v16bf pa = load_a_frag(Pw, 0, DH, ks * 32, lane);
#pragma unroll
      for (int j = 0; j < 4; ++j) {
        v16bf vb = load_b_frag(Vt[cur], j * 16, BK, ks * 32, lane);
        o[j] = WMMA_BF16(pa, vb, o[j]);
      }
    }

    if (kbi + 1 < NKB) s_store(cur ^ 1);
  }

  // normalize and write head-concat output (bf16, [B*S, D], head at col h*64)
  bf16* Og = Out + (size_t)(b * S + q0 + qr) * Dm + h * DH;
#pragma unroll
  for (int e = 0; e < 8; ++e) {
    float inv = 1.0f / l[e];
#pragma unroll
    for (int j = 0; j < 4; ++j)
      Og[(size_t)(hh * 8 + e) * Dm + j * 16 + ln] = (bf16)(o[j][e] * inv);
  }
}

// ---------------------------------------------------------------------------

extern "C" void kernel_launch(void* const* d_in, const int* in_sizes, int n_in,
                              void* d_out, int out_size, void* d_ws,
                              size_t ws_size, hipStream_t stream) {
  const float* q  = (const float*)d_in[0];
  const float* k  = (const float*)d_in[1];
  const float* v  = (const float*)d_in[2];
  const float* Wq = (const float*)d_in[3];
  const float* Wk = (const float*)d_in[4];
  const float* Wv = (const float*)d_in[5];
  const float* Wo = (const float*)d_in[6];

  const size_t NR = 4096ull * 1024ull;   // B*S x D elements
  bf16* Qp = (bf16*)d_ws;                // 8 MB each, 33.5 MB total
  bf16* Kp = Qp + NR;
  bf16* Vp = Kp + NR;
  bf16* Ao = Vp + NR;

  dim3 blk(256);
  dim3 gridProj(8, 32);   // N-tiles (1024/128), M-tiles (4096/128)
  // Fold log2(e)/sqrt(DH) into the Q projection: softmax then runs in the
  // base-2 domain, so the flash kernel uses raw v_exp_f32 (no log2e muls).
  const float qscale = 0.125f * 1.44269504088896340736f;
  gemm_nt_kernel<false, true><<<gridProj, blk, 0, stream>>>(q, Wq, Qp, qscale);
  gemm_nt_kernel<false, true><<<gridProj, blk, 0, stream>>>(k, Wk, Kp, 1.0f);
  gemm_nt_kernel<false, true><<<gridProj, blk, 0, stream>>>(v, Wv, Vp, 1.0f);

  dim3 gridAttn(16, 32);  // q-tiles (2048/128), B*H
  flash_attn_kernel<<<gridAttn, blk, 0, stream>>>(Qp, Kp, Vp, Ao);

  gemm_nt_kernel<true, false><<<gridProj, blk, 0, stream>>>(Ao, Wo, d_out, 1.0f);
}
